// SelfAttention_12678743458584
// MI455X (gfx1250) — matrix operand order repaired and checked by reading, not compiled, not run on hardware
//
#include <hip/hip_runtime.h>

typedef __attribute__((ext_vector_type(16))) __bf16 v16bf;
typedef __attribute__((ext_vector_type(8)))  __bf16 v8bf;
typedef __attribute__((ext_vector_type(8)))  float  v8f;

#define BATCH 4
#define SEQ   2048
#define DEMB  1024
#define NHEAD 16
#define DHEAD 64
#define ROWS  (BATCH*SEQ)   /* 8192 */
#define E3D   (3*DEMB)      /* 3072 */

__device__ __forceinline__ unsigned short f32_to_bf16(float f) {
  unsigned int u = __float_as_uint(f);
  u += 0x7FFFu + ((u >> 16) & 1u);           // round-to-nearest-even
  return (unsigned short)(u >> 16);
}

// A-operand (16x32 bf16): lane = M; lane-half selects contiguous K half (k = 16*sel + s).
// Two 16B loads: bytes [sel*32, sel*32+16) and [sel*32+16, sel*32+32) of the row.
__device__ __forceinline__ v16bf load_a16(const __bf16* base, int ld, int lane) {
  const int m   = lane & 15;
  const int sel = (lane >> 4) & 1;
  const __bf16* p = base + m * ld + sel * 16;
  union { v16bf v; v8bf h[2]; } u;
  u.h[0] = *(const v8bf*)(p);
  u.h[1] = *(const v8bf*)(p + 8);
  return u.v;
}

// B-operand (32x16 bf16): lane = N, lane-half selects K half; memory is [N, K]
// row-major (K contiguous) -> one contiguous 32B load per lane.
__device__ __forceinline__ v16bf load_b16(const __bf16* base, int ld, int lane) {
  const int n   = lane & 15;
  const int sel = (lane >> 4) & 1;
  return *(const v16bf*)(base + n * ld + sel * 16);
}

__device__ __forceinline__ v8f wmma_bf16(v16bf a, v16bf b, v8f c) {
  return __builtin_amdgcn_wmma_f32_16x16x32_bf16(false, a, false, b,
                                                 (short)0, c, false, false);
}

/* ---------------- f32 -> bf16 conversion (4 elems / thread) ---------------- */
__global__ void __launch_bounds__(256)
cvt_f32_bf16(const float* __restrict__ in, unsigned short* __restrict__ out, int n4) {
  int i = blockIdx.x * blockDim.x + threadIdx.x;
  if (i < n4) {
    float4 f = ((const float4*)in)[i];
    unsigned int lo = (unsigned int)f32_to_bf16(f.x) | ((unsigned int)f32_to_bf16(f.y) << 16);
    unsigned int hi = (unsigned int)f32_to_bf16(f.z) | ((unsigned int)f32_to_bf16(f.w) << 16);
    ((uint2*)out)[i] = make_uint2(lo, hi);
  }
}

/* ------------- QKV GEMM: [8192,3072] = Xb * Winb^T, scatter to Q/K/Vt ------
   Mapping: the 8 waves of a block share the same 64-wide N tile (same tn) so
   the B operand is served from WGP$ once per block. */
__global__ void __launch_bounds__(256)
qkv_gemm(const unsigned short* __restrict__ xb,
         const unsigned short* __restrict__ winb,
         const float* __restrict__ b_in,
         unsigned short* __restrict__ q,
         unsigned short* __restrict__ k,
         unsigned short* __restrict__ vt) {
  const int lane = threadIdx.x & 31;
  const int wave = threadIdx.x >> 5;
  const int wid  = blockIdx.x * 8 + wave;      // 3072 blocks * 8 waves = 24576
  const int tm   = wid & 511;                  // M tile of 16 (varies within block)
  const int tn   = wid >> 9;                   // N tile of 64 (uniform per block)

  const __bf16* A  = (const __bf16*)xb   + (size_t)tm * 16 * DEMB;
  const __bf16* Bw = (const __bf16*)winb + (size_t)tn * 64 * DEMB;

  v8f acc0 = {}, acc1 = {}, acc2 = {}, acc3 = {};
  for (int kk = 0; kk < DEMB; kk += 32) {
    v16bf a = load_a16(A + kk, DEMB, lane);
    acc0 = wmma_bf16(a, load_b16(Bw + 0 * 16 * DEMB + kk, DEMB, lane), acc0);
    acc1 = wmma_bf16(a, load_b16(Bw + 1 * 16 * DEMB + kk, DEMB, lane), acc1);
    acc2 = wmma_bf16(a, load_b16(Bw + 2 * 16 * DEMB + kk, DEMB, lane), acc2);
    acc3 = wmma_bf16(a, load_b16(Bw + 3 * 16 * DEMB + kk, DEMB, lane), acc3);
  }

  v8f acc[4] = {acc0, acc1, acc2, acc3};
  const int mbase = tm * 16;
  const int rb    = 8 * (lane >> 4);
  const int col   = lane & 15;
#pragma unroll
  for (int j = 0; j < 4; ++j) {
    const int e     = tn * 64 + j * 16 + col;
    const float bia = b_in[e];
    const int which = e >> 10;                 // 0=Q 1=K 2=V
    const int h     = (e >> 6) & 15;
    const int d     = e & 63;
#pragma unroll
    for (int r = 0; r < 8; ++r) {
      const int row = mbase + r + rb;
      const int b   = row >> 11;
      const int s   = row & (SEQ - 1);
      const int bh  = b * NHEAD + h;
      float v = acc[j][r] + bia;
      if (which == 0) {
        // fold the softmax 1/Dh score scale into Q before bf16 rounding
        q[((size_t)bh * SEQ + s) * DHEAD + d] = f32_to_bf16(v * (1.0f / 64.0f));
      } else if (which == 1) {
        k[((size_t)bh * SEQ + s) * DHEAD + d] = f32_to_bf16(v);
      } else {
        vt[((size_t)bh * DHEAD + d) * SEQ + s] = f32_to_bf16(v);  // V transposed
      }
    }
  }
}

/* ------------- Flash attention: one wave per (b,h, 16-query tile) ----------
   128 keys per iteration: 8 score tiles in registers, one softmax pass, one
   16x128 bf16 P tile through LDS, 16 PV WMMAs. Block waves share K/V via WGP$. */
__global__ void __launch_bounds__(256)
attn_flash(const unsigned short* __restrict__ qg,
           const unsigned short* __restrict__ kg,
           const unsigned short* __restrict__ vtg,
           unsigned short* __restrict__ ao) {
  __shared__ unsigned short pbuf[8 * 16 * 128];         // per-wave 16x128 bf16 P tile
  const int lane = threadIdx.x & 31;
  const int wave = threadIdx.x >> 5;
  const int wid  = blockIdx.x * 8 + wave;               // 1024 blocks * 8 = 8192
  const int bh   = wid >> 7;                            // 0..63 (uniform per block)
  const int qt   = wid & 127;                           // query tile of 16

  const __bf16* Q = (const __bf16*)qg  + ((size_t)bh * SEQ + qt * 16) * DHEAD;
  const __bf16* K = (const __bf16*)kg  + (size_t)bh * SEQ * DHEAD;
  const __bf16* V = (const __bf16*)vtg + (size_t)bh * DHEAD * SEQ;
  unsigned short* myp = &pbuf[wave * 16 * 128];

  const v16bf aq0 = load_a16(Q + 0,  DHEAD, lane);      // K dims 0..31 (pre-scaled)
  const v16bf aq1 = load_a16(Q + 32, DHEAD, lane);      // K dims 32..63

  v8f o0 = {}, o1 = {}, o2 = {}, o3 = {};
  float mrow[8], srow[8];
#pragma unroll
  for (int r = 0; r < 8; ++r) { mrow[r] = -3.0e38f; srow[r] = 0.0f; }

  const int rb  = 8 * (lane >> 4);
  const int col = lane & 15;

  for (int kt = 0; kt < SEQ; kt += 128) {
    // scores for 128 keys: 8 16x16 C tiles (16 WMMAs)
    v8f c[8];
#pragma unroll
    for (int t = 0; t < 8; ++t) {
      v8f z = {};
      z = wmma_bf16(aq0, load_b16(K + (size_t)(kt + t * 16) * DHEAD +  0, DHEAD, lane), z);
      z = wmma_bf16(aq1, load_b16(K + (size_t)(kt + t * 16) * DHEAD + 32, DHEAD, lane), z);
      c[t] = z;
    }

    // one online-softmax pass over the 128-key strip, per row r
#pragma unroll
    for (int r = 0; r < 8; ++r) {
      float s[8];
      float mx = -3.0e38f;
#pragma unroll
      for (int t = 0; t < 8; ++t) { s[t] = c[t][r]; mx = fmaxf(mx, s[t]); }
      mx = fmaxf(mx, __shfl_xor(mx, 1, 32));
      mx = fmaxf(mx, __shfl_xor(mx, 2, 32));
      mx = fmaxf(mx, __shfl_xor(mx, 4, 32));
      mx = fmaxf(mx, __shfl_xor(mx, 8, 32));
      const float mnew = fmaxf(mrow[r], mx);
      const float corr = __expf(mrow[r] - mnew);
      float ps = 0.0f;
#pragma unroll
      for (int t = 0; t < 8; ++t) { s[t] = __expf(s[t] - mnew); ps += s[t]; }
      ps += __shfl_xor(ps, 1, 32);
      ps += __shfl_xor(ps, 2, 32);
      ps += __shfl_xor(ps, 4, 32);
      ps += __shfl_xor(ps, 8, 32);
      srow[r] = srow[r] * corr + ps;
      mrow[r] = mnew;
      o0[r] *= corr; o1[r] *= corr; o2[r] *= corr; o3[r] *= corr;
      // C-layout -> LDS row-major 16x128 bf16 P tile
#pragma unroll
      for (int t = 0; t < 8; ++t)
        myp[(r + rb) * 128 + t * 16 + col] = f32_to_bf16(s[t]);
    }
    asm volatile("s_wait_dscnt 0" ::: "memory");

    // O += P * V over K=128 in four K=32 chunks (V pre-transposed: [d, s])
#pragma unroll
    for (int kb = 0; kb < 4; ++kb) {
      const v16bf ap = load_a16((const __bf16*)myp + kb * 32, 128, lane);
      const int kc = kt + kb * 32;
      o0 = wmma_bf16(ap, load_b16(V + (size_t)( 0) * SEQ + kc, SEQ, lane), o0);
      o1 = wmma_bf16(ap, load_b16(V + (size_t)(16) * SEQ + kc, SEQ, lane), o1);
      o2 = wmma_bf16(ap, load_b16(V + (size_t)(32) * SEQ + kc, SEQ, lane), o2);
      o3 = wmma_bf16(ap, load_b16(V + (size_t)(48) * SEQ + kc, SEQ, lane), o3);
    }
  }

  // normalize and store to attn-out [B, S, H*Dh] (bf16)
  const int b = bh >> 4, h = bh & 15;
  v8f oo[4] = {o0, o1, o2, o3};
#pragma unroll
  for (int r = 0; r < 8; ++r) {
    const float inv = 1.0f / srow[r];
    const size_t rowoff = ((size_t)b * SEQ + qt * 16 + r + rb) * DEMB + h * DHEAD;
#pragma unroll
    for (int j = 0; j < 4; ++j)
      ao[rowoff + j * 16 + col] = f32_to_bf16(oo[j][r] * inv);
  }
}

/* ------------- Output projection: [8192,1024] f32 = AO * Wout^T + b -------- */
__global__ void __launch_bounds__(256)
out_gemm(const unsigned short* __restrict__ ao,
         const unsigned short* __restrict__ woutb,
         const float* __restrict__ b_out,
         float* __restrict__ out) {
  const int lane = threadIdx.x & 31;
  const int wave = threadIdx.x >> 5;
  const int wid  = blockIdx.x * 8 + wave;      // 1024 blocks * 8 = 8192
  const int tm   = wid & 511;                  // M tile (varies within block)
  const int tn   = wid >> 9;                   // N tile (uniform per block)

  const __bf16* A  = (const __bf16*)ao    + (size_t)tm * 16 * DEMB;
  const __bf16* Bw = (const __bf16*)woutb + (size_t)tn * 64 * DEMB;

  v8f acc0 = {}, acc1 = {}, acc2 = {}, acc3 = {};
  for (int kk = 0; kk < DEMB; kk += 32) {
    v16bf a = load_a16(A + kk, DEMB, lane);
    acc0 = wmma_bf16(a, load_b16(Bw + 0 * 16 * DEMB + kk, DEMB, lane), acc0);
    acc1 = wmma_bf16(a, load_b16(Bw + 1 * 16 * DEMB + kk, DEMB, lane), acc1);
    acc2 = wmma_bf16(a, load_b16(Bw + 2 * 16 * DEMB + kk, DEMB, lane), acc2);
    acc3 = wmma_bf16(a, load_b16(Bw + 3 * 16 * DEMB + kk, DEMB, lane), acc3);
  }

  v8f acc[4] = {acc0, acc1, acc2, acc3};
  const int rb  = 8 * (lane >> 4);
  const int col = lane & 15;
#pragma unroll
  for (int j = 0; j < 4; ++j) {
    const int c   = tn * 64 + j * 16 + col;
    const float bia = b_out[c];
#pragma unroll
    for (int r = 0; r < 8; ++r) {
      const int row = tm * 16 + r + rb;
      out[(size_t)row * DEMB + c] = acc[j][r] + bia;
    }
  }
}

/* --------------------------------- launch ---------------------------------- */
extern "C" void kernel_launch(void* const* d_in, const int* in_sizes, int n_in,
                              void* d_out, int out_size, void* d_ws, size_t ws_size,
                              hipStream_t stream) {
  const float* x     = (const float*)d_in[0];   // [4,2048,1024]
  const float* w_in  = (const float*)d_in[1];   // [3072,1024]
  const float* b_in  = (const float*)d_in[2];   // [3072]
  const float* w_out = (const float*)d_in[3];   // [1024,1024]
  const float* b_out = (const float*)d_in[4];   // [1024]
  float* out = (float*)d_out;                   // [4,2048,1024]

  unsigned short* xb    = (unsigned short*)d_ws;
  unsigned short* winb  = xb    + (size_t)ROWS * DEMB;
  unsigned short* woutb = winb  + (size_t)E3D  * DEMB;
  unsigned short* qd    = woutb + (size_t)DEMB * DEMB;
  unsigned short* kd    = qd    + (size_t)ROWS * DEMB;
  unsigned short* vtd   = kd    + (size_t)ROWS * DEMB;
  unsigned short* aod   = vtd   + (size_t)ROWS * DEMB;

  const int nx = ROWS * DEMB / 4, nwi = E3D * DEMB / 4, nwo = DEMB * DEMB / 4;
  cvt_f32_bf16<<<(nx  + 255) / 256, 256, 0, stream>>>(x,     xb,    nx);
  cvt_f32_bf16<<<(nwi + 255) / 256, 256, 0, stream>>>(w_in,  winb,  nwi);
  cvt_f32_bf16<<<(nwo + 255) / 256, 256, 0, stream>>>(w_out, woutb, nwo);

  qkv_gemm  <<<(512 * (E3D / 64)) / 8,  256, 0, stream>>>(xb, winb, b_in, qd, kd, vtd);
  attn_flash<<<(64 * 128) / 8,          256, 0, stream>>>(qd, kd, vtd, aod);
  out_gemm  <<<(512 * (DEMB / 64)) / 8, 256, 0, stream>>>(aod, woutb, b_out, out);

  (void)in_sizes; (void)n_in; (void)out_size; (void)ws_size;
}
